// Head_85066122264693
// MI455X (gfx1250) — matrix operand order repaired and checked by reading, not compiled, not run on hardware
//
#include <hip/hip_runtime.h>
#include <hip/hip_bf16.h>

typedef __bf16 bf16;
typedef __attribute__((ext_vector_type(16))) __bf16 v16bf;
typedef __attribute__((ext_vector_type(8)))  __bf16 v8bf;
typedef __attribute__((ext_vector_type(8)))  float  v8f;

#define B_  4
#define T_  2048
#define C_  1024
#define H_  64
#define BT_ 8192

static __device__ __forceinline__ v8f wmma_bf16(v16bf a, v16bf b, v8f c) {
  return __builtin_amdgcn_wmma_f32_16x16x32_bf16(false, a, false, b, (short)0, c, false, false);
}

// ---------------------------------------------------------------------------
// Phase 0: convert W{q,k,v} (fp32 [C,H]) to bf16 transposed Wt [3][H][C].
// One-shot, tiny; makes phase-1 B-fragment loads fully contiguous.
// ---------------------------------------------------------------------------
__global__ __launch_bounds__(256) void prep_w_kernel(
    const float* __restrict__ Wq, const float* __restrict__ Wk,
    const float* __restrict__ Wv, bf16* __restrict__ Wt) {
  int tid = blockIdx.x * 256 + threadIdx.x;           // 0 .. 3*C*H-1
  if (tid >= 3 * C_ * H_) return;
  const int mat = tid / (C_ * H_);
  const int r   = tid % (C_ * H_);
  const int c   = r / H_;
  const int h   = r % H_;
  const float* __restrict__ W = (mat == 0) ? Wq : ((mat == 1) ? Wk : Wv);
  Wt[((size_t)mat * H_ + h) * C_ + c] = (bf16)W[(size_t)c * H_ + h];
}

// ---------------------------------------------------------------------------
// Phase 1: Q = x@Wq, K = x@Wk, V = x@Wv  (fp32 x -> bf16, fp32-accum WMMA)
// One wave per (16-row M tile, matrix). 512 M tiles * 3 matrices = 1536 waves.
// V is stored transposed [H, BT] so phase 2 reads contiguous B-fragments.
// ---------------------------------------------------------------------------
__global__ __launch_bounds__(128) void qkv_proj_kernel(
    const float* __restrict__ x, const bf16* __restrict__ Wt,
    bf16* __restrict__ Qb, bf16* __restrict__ Kb, bf16* __restrict__ Vt) {
  const int lane = threadIdx.x & 31;
  // wid/mat/mtile are wave-uniform: force them into SGPRs for scalar branches
  const int wid =
      __builtin_amdgcn_readfirstlane(blockIdx.x * 4 + (threadIdx.x >> 5));
  const int mat   = wid % 3;       // 0=Q 1=K 2=V
  const int row0  = (wid / 3) * 16;
  const int ln = lane & 15, hi = lane >> 4;

  const float* __restrict__ xrow = x + (size_t)(row0 + ln) * C_;
  const bf16* __restrict__ wbase = Wt + (size_t)mat * H_ * C_;

  v8f acc[4] = {v8f{}, v8f{}, v8f{}, v8f{}};

  for (int kb = 0; kb < C_; kb += 32) {
    // A fragment: x[row0+ln][kb..kb+32), contiguous K order (K = e + 16*hi)
    v16bf a;
    const float* pa = xrow + kb;
#pragma unroll
    for (int e = 0; e < 8; e++) a[e]     = (bf16)pa[16 * hi + e];      // K = e + 16*hi
#pragma unroll
    for (int e = 0; e < 8; e++) a[e + 8] = (bf16)pa[16 * hi + 8 + e];  // K = 8+e+16*hi

#pragma unroll
    for (int t = 0; t < 4; t++) {
      // B fragment: Wt row = (16t+ln), cols kb+16*hi .. +16 : contiguous 32B
      v16bf bf = *(const v16bf*)(wbase + (size_t)(16 * t + ln) * C_ + kb + 16 * hi);
      acc[t] = wmma_bf16(a, bf, acc[t]);
    }
  }

  // Convert C/D tile to bf16 once (VGPR r holds row r+8*hi, column ln)
  v8bf vals[4];
#pragma unroll
  for (int t = 0; t < 4; t++)
#pragma unroll
    for (int r = 0; r < 8; r++) vals[t][r] = (bf16)acc[t][r];

  if (mat == 2) {
    // Vt[h][m]: for fixed lane (h = 16t+ln), rows r are CONTIGUOUS -> b128
#pragma unroll
    for (int t = 0; t < 4; t++) {
      bf16* vp = Vt + (size_t)(16 * t + ln) * BT_ + row0 + 8 * hi;
      *(v8bf*)vp = vals[t];
    }
  } else {
    bf16* __restrict__ dst = (mat == 0) ? Qb : Kb;
#pragma unroll
    for (int t = 0; t < 4; t++)
#pragma unroll
      for (int r = 0; r < 8; r++)
        dst[(size_t)(row0 + r + 8 * hi) * H_ + 16 * t + ln] = vals[t][r];
  }
}

// ---------------------------------------------------------------------------
// Phase 2: flash attention. One wave per 16-query tile (512 waves).
// Key block = 32 (two 16x16 S tiles). Online softmax; P relayout via LDS.
// ---------------------------------------------------------------------------
__global__ __launch_bounds__(128) void attn_kernel(
    const bf16* __restrict__ Qb, const bf16* __restrict__ Kb,
    const bf16* __restrict__ Vt, const unsigned char* __restrict__ maskp,
    float* __restrict__ out) {
  __shared__ bf16 plds[4][16 * 32];   // per-wave 16x32 bf16 P tile

  const int widx = threadIdx.x >> 5;
  const int lane = threadIdx.x & 31;
  const int wid  = __builtin_amdgcn_readfirstlane(blockIdx.x * 4 + widx);
  const int b    = wid >> 7;                // batch       (scalar)
  const int qb   = (wid & 127) * 16;        // q tile base (scalar)
  const int ln = lane & 15, hi = lane >> 4;
  const float scale = 0.03125f;             // 1/sqrt(C)

  // Q A-fragments for the two 32-wide H chunks
  v16bf aq[2];
  const bf16* qrow = Qb + (size_t)(b * T_ + qb + ln) * H_;
#pragma unroll
  for (int c = 0; c < 2; c++) {
#pragma unroll
    for (int e = 0; e < 8; e++) {
      aq[c][e]     = qrow[c * 32 + 16 * hi + e];
      aq[c][e + 8] = qrow[c * 32 + 16 * hi + 8 + e];
    }
  }

  float mrow[8], lrow[8];
  v8f acc[4] = {v8f{}, v8f{}, v8f{}, v8f{}};
#pragma unroll
  for (int r = 0; r < 8; r++) { mrow[r] = -3.0e38f; lrow[r] = 0.0f; }

  const int jmax = (qb + 15) >> 5;          // causal: last key block index
  for (int j = 0; j <= jmax; j++) {
    const int kb = j * 32;

    const bf16* krow0 = Kb + (size_t)(b * T_ + kb + ln) * H_;
    const bf16* krow1 = krow0 + (size_t)16 * H_;
    // Prefetch next key block's K rows (global_prefetch_b8, speculative)
    __builtin_prefetch(krow0 + (size_t)32 * H_, 0, 3);
    __builtin_prefetch(krow1 + (size_t)32 * H_, 0, 3);

    // S = Q @ K^T : two 16x16 tiles (key cols kb+ln and kb+16+ln)
    v8f S0 = v8f{}, S1 = v8f{};
#pragma unroll
    for (int c = 0; c < 2; c++) {
      v16bf b0 = *(const v16bf*)(krow0 + c * 32 + 16 * hi);
      v16bf b1 = *(const v16bf*)(krow1 + c * 32 + 16 * hi);
      S0 = wmma_bf16(aq[c], b0, S0);
      S1 = wmma_bf16(aq[c], b1, S1);
    }

    const int key0 = kb + ln, key1 = kb + 16 + ln;
    const bool v0 = maskp[(size_t)b * T_ + key0] != 0;
    const bool v1 = maskp[(size_t)b * T_ + key1] != 0;

    float p0[8], p1[8];
#pragma unroll
    for (int r = 0; r < 8; r++) {
      const int qt = qb + r + 8 * hi;       // global query row (in batch)
      const float s0 = (v0 && key0 <= qt) ? S0[r] * scale : -1.0e30f;
      const float s1 = (v1 && key1 <= qt) ? S1[r] * scale : -1.0e30f;
      float mx = fmaxf(s0, s1);
      mx = fmaxf(mx, __shfl_xor(mx, 1));
      mx = fmaxf(mx, __shfl_xor(mx, 2));
      mx = fmaxf(mx, __shfl_xor(mx, 4));
      mx = fmaxf(mx, __shfl_xor(mx, 8));    // row-max within 16-lane half
      const float mnew  = fmaxf(mrow[r], mx);
      const float alpha = __expf(mrow[r] - mnew);
      mrow[r] = mnew;
      lrow[r] *= alpha;
      acc[0][r] *= alpha; acc[1][r] *= alpha;
      acc[2][r] *= alpha; acc[3][r] *= alpha;
      const float e0 = __expf(s0 - mnew);
      const float e1 = __expf(s1 - mnew);
      float sum = e0 + e1;
      sum += __shfl_xor(sum, 1);
      sum += __shfl_xor(sum, 2);
      sum += __shfl_xor(sum, 4);
      sum += __shfl_xor(sum, 8);
      lrow[r] += sum;
      p0[r] = e0; p1[r] = e1;
    }

    // P: C/D layout -> LDS (row-major 16x32 bf16) -> A layout.
    // Same-wave RAW/WAR only; block barrier is ILLEGAL here (divergent jmax).
    bf16* pl = plds[widx];
#pragma unroll
    for (int r = 0; r < 8; r++) {
      const int row = r + 8 * hi;
      pl[row * 32 + ln]      = (bf16)p0[r];
      pl[row * 32 + 16 + ln] = (bf16)p1[r];
    }
    asm volatile("s_wait_dscnt 0" ::: "memory");

    v16bf ap;
    const bf16* pr = pl + ln * 32;
#pragma unroll
    for (int e = 0; e < 8; e++) {
      ap[e]     = pr[16 * hi + e];
      ap[e + 8] = pr[16 * hi + 8 + e];
    }
    asm volatile("s_wait_dscnt 0" ::: "memory");

    // O += P @ V : V^T gives contiguous B fragments (k = e + 16*hi)
#pragma unroll
    for (int t = 0; t < 4; t++) {
      const bf16* vp = Vt + (size_t)(16 * t + ln) * BT_ + b * T_ + kb + 16 * hi;
      v16bf bv = *(const v16bf*)vp;
      acc[t] = wmma_bf16(ap, bv, acc[t]);
    }
  }

  // out[b, q, h] = acc / l
#pragma unroll
  for (int t = 0; t < 4; t++)
#pragma unroll
    for (int r = 0; r < 8; r++) {
      const int qt = qb + r + 8 * hi;
      out[(size_t)(b * T_ + qt) * H_ + 16 * t + ln] = acc[t][r] / lrow[r];
    }
}

extern "C" void kernel_launch(void* const* d_in, const int* in_sizes, int n_in,
                              void* d_out, int out_size, void* d_ws, size_t ws_size,
                              hipStream_t stream) {
  const float*         x     = (const float*)d_in[0];
  const unsigned char* maskp = (const unsigned char*)d_in[1];  // bool [B,T]
  const float*         Wq    = (const float*)d_in[2];
  const float*         Wk    = (const float*)d_in[3];
  const float*         Wv    = (const float*)d_in[4];
  float*               out   = (float*)d_out;

  bf16* Qb = (bf16*)d_ws;              // [BT, H]
  bf16* Kb = Qb + (size_t)BT_ * H_;    // [BT, H]
  bf16* Vt = Kb + (size_t)BT_ * H_;    // [H, BT]
  bf16* Wt = Vt + (size_t)BT_ * H_;    // [3, H, C] transposed bf16 weights

  // Phase 0: one-shot weight convert/transpose (3*C*H = 196608 elems)
  prep_w_kernel<<<(3 * C_ * H_ + 255) / 256, 256, 0, stream>>>(Wq, Wk, Wv, Wt);
  // Phase 1: 1536 waves -> 384 blocks x 128 threads (4 waves/block)
  qkv_proj_kernel<<<384, 128, 0, stream>>>(x, Wt, Qb, Kb, Vt);
  // Phase 2: 512 waves -> 128 blocks x 128 threads
  attn_kernel<<<128, 128, 0, stream>>>(Qb, Kb, Vt, maskp, out);
}
